// RoleSpecificMultiTaskGNN_86311662781033
// MI455X (gfx1250) — matrix operand order repaired and checked by reading, not compiled, not run on hardware
//
#include <hip/hip_runtime.h>

typedef __attribute__((ext_vector_type(16))) __bf16  v16bf;
typedef __attribute__((ext_vector_type(2)))  __bf16  v2bf;
typedef __attribute__((ext_vector_type(8)))  float   v8f;

#define BM 128
#define BN 128
#define BK 32
#define KP 40   // padded LDS K-stride (ushorts); 80B rows keep 4B alignment for pair loads

__device__ __forceinline__ unsigned short f2bf(float f) {
  unsigned u = __builtin_bit_cast(unsigned, f);
  return (unsigned short)((u + 0x7FFFu + ((u >> 16) & 1u)) >> 16);   // RNE f32->bf16
}

// pack two f32 -> two bf16 in one dword (prefer HW packed convert if available)
__device__ __forceinline__ unsigned pk2bf(float lo, float hi) {
#if __has_builtin(__builtin_amdgcn_cvt_pk_bf16_f32)
  v2bf r = __builtin_amdgcn_cvt_pk_bf16_f32(lo, hi);
  return __builtin_bit_cast(unsigned, r);
#else
  return (unsigned)f2bf(lo) | ((unsigned)f2bf(hi) << 16);
#endif
}

union FragBF { unsigned u[8]; v16bf v; };

// C[M,256] = A1[M,K1] @ B1[K1,256]  (+ A2[M,K2] @ B2[K2,256] if A2 != null)
// A is f32 row-major (stride = K), converted to bf16 while staging into LDS.
// Block tile 128x128 (grid.x = 2 for the fixed 256-wide N), 8 waves, wave tile 32x64.
__global__ __launch_bounds__(256) void gemm_bf16_k(
    const float* __restrict__ A1, const float* __restrict__ B1, int K1,
    const float* __restrict__ A2, const float* __restrict__ B2, int K2,
    float* __restrict__ C, int Mtot)
{
  __shared__ unsigned short As[BM * KP];   // [m][k] bf16
  __shared__ unsigned short Bs[BN * KP];   // [n][k] bf16 (transposed stage)
  const int tid  = threadIdx.x;
  const int lane = tid & 31;
  const int wid  = tid >> 5;
  const int wm   = (wid & 3) * 32;     // 4 waves along M
  const int wn   = (wid >> 2) * 64;    // 2 waves along N
  const int m0   = blockIdx.y * BM;
  const int n0   = blockIdx.x * BN;

  v8f acc[2][4] = {};

  for (int seg = 0; seg < 2; ++seg) {
    const float* __restrict__ A = seg ? A2 : A1;
    const float* __restrict__ B = seg ? B2 : B1;
    const int K = seg ? K2 : K1;
    if (A == nullptr || K == 0) continue;

    for (int k0 = 0; k0 < K; k0 += BK) {
      // stage A tile: 128x32 f32 -> bf16 (1024 float4 tasks, 4 per thread)
      #pragma unroll
      for (int it = 0; it < 4; ++it) {
        int f   = tid + it * 256;
        int row = f >> 3;
        int kc  = (f & 7) << 2;
        float4 val = make_float4(0.f, 0.f, 0.f, 0.f);
        int gr = m0 + row;
        if (gr < Mtot) {
          const float* ap = A + (size_t)gr * K + k0 + kc;
          val = *(const float4*)ap;
          if (k0 + BK < K) __builtin_prefetch(ap + BK, 0, 3);  // next k-tile
        }
        unsigned* p = (unsigned*)&As[row * KP + kc];
        p[0] = pk2bf(val.x, val.y);
        p[1] = pk2bf(val.z, val.w);
      }
      // stage B tile transposed with k-pair packing:
      // 512 tasks (16 k-pairs x 32 n-float4), 2 per thread
      #pragma unroll
      for (int it = 0; it < 2; ++it) {
        int f  = tid + it * 256;
        int kp = f >> 5;             // k-pair index 0..15
        int nc = (f & 31) << 2;      // n offset 0..124
        const float* bp = B + (size_t)(k0 + kp * 2) * 256 + n0 + nc;
        float4 r0 = *(const float4*)bp;
        float4 r1 = *(const float4*)(bp + 256);
        unsigned short* q = &Bs[nc * KP + kp * 2];
        *(unsigned*)(q + 0 * KP) = pk2bf(r0.x, r1.x);
        *(unsigned*)(q + 1 * KP) = pk2bf(r0.y, r1.y);
        *(unsigned*)(q + 2 * KP) = pk2bf(r0.z, r1.z);
        *(unsigned*)(q + 3 * KP) = pk2bf(r0.w, r1.w);
      }
      __syncthreads();
      {
        const int l15 = lane & 15;
        const int kh  = lane >> 4;
        FragBF a[2], b[4];
        #pragma unroll
        for (int t = 0; t < 2; ++t) {
          // A 16x32 bf16 layout: lane m=l15, half kh; VGPR v covers K = 2v (+8 if v>=4) + 8*kh
          const unsigned short* ap = &As[(wm + t * 16 + l15) * KP + kh * 8];
          #pragma unroll
          for (int v = 0; v < 8; ++v) {
            int ko = 2 * v + ((v >= 4) ? 8 : 0);
            a[t].u[v] = *(const unsigned*)(ap + ko);
          }
        }
        #pragma unroll
        for (int t = 0; t < 4; ++t) {
          // B 32x16 bf16 layout: lane n=l15, half kh; VGPR v covers K = 2v (+8 if v>=4) + 8*kh
          const unsigned short* bp = &Bs[(wn + t * 16 + l15) * KP + kh * 8];
          #pragma unroll
          for (int v = 0; v < 8; ++v)
            b[t].u[v] = *(const unsigned*)(bp + 2 * v + ((v >= 4) ? 8 : 0));
        }
        #pragma unroll
        for (int ti = 0; ti < 2; ++ti)
          #pragma unroll
          for (int tj = 0; tj < 4; ++tj)
            acc[ti][tj] = __builtin_amdgcn_wmma_f32_16x16x32_bf16(
                false, a[ti].v, false, b[tj].v, (short)0, acc[ti][tj], false, false);
      }
      __syncthreads();
    }
  }

  // C/D layout: n = lane&15, m = v + 8*(lane>>4)
  const int nn = lane & 15;
  const int mh = (lane >> 4) * 8;
  #pragma unroll
  for (int ti = 0; ti < 2; ++ti)
    #pragma unroll
    for (int tj = 0; tj < 4; ++tj)
      #pragma unroll
      for (int v = 0; v < 8; ++v) {
        int gr = m0 + wm + ti * 16 + mh + v;
        int gc = n0 + wn + tj * 16 + nn;
        if (gr < Mtot) C[(size_t)gr * 256 + gc] = acc[ti][tj][v];
      }
}

__global__ __launch_bounds__(256) void zero_k(float* __restrict__ p, int n) {
  int idx = (blockIdx.x * 256 + threadIdx.x) * 4;
  if (idx < n) *(float4*)(p + idx) = make_float4(0.f, 0.f, 0.f, 0.f);
}

// per edge: sums[dst*1280 + et*256 + h] += x[src*256 + h];  cnt[dst*5 + et] += 1
__global__ __launch_bounds__(256) void scatter_k(
    const float* __restrict__ x, const int* __restrict__ src, const int* __restrict__ dst,
    const int* __restrict__ et, float* __restrict__ sums, float* __restrict__ cnt, int E)
{
  int e = blockIdx.x * 4 + (threadIdx.x >> 6);
  if (e >= E) return;
  int h4 = (threadIdx.x & 63) << 2;
  int s = src[e], d = dst[e], t = et[e];
  float4 val = *(const float4*)(x + (size_t)s * 256 + h4);
  float* p = sums + (size_t)d * 1280 + t * 256 + h4;
  atomicAdd(p + 0, val.x);
  atomicAdd(p + 1, val.y);
  atomicAdd(p + 2, val.z);
  atomicAdd(p + 3, val.w);
  if (h4 == 0) atomicAdd(cnt + d * 5 + t, 1.0f);
}

__global__ __launch_bounds__(256) void meandiv_k(
    float* __restrict__ sums, const float* __restrict__ cnt, int total)
{
  int idx = blockIdx.x * 256 + threadIdx.x;
  if (idx >= total) return;
  int n = idx / 1280;
  int r = (idx - n * 1280) >> 8;
  float k = cnt[n * 5 + r];
  sums[idx] = sums[idx] / fmaxf(k, 1.0f);
}

// out = [relu]( LN(y + bias) * g + b ) [+ res] ; one wave per 256-wide row
__global__ __launch_bounds__(256) void ln_k(
    const float* __restrict__ y, const float* __restrict__ bias,
    const float* __restrict__ g, const float* __restrict__ b,
    const float* __restrict__ res, float* __restrict__ out, int M, int do_relu)
{
  int row  = blockIdx.x * 8 + (threadIdx.x >> 5);
  int lane = threadIdx.x & 31;
  if (row >= M) return;
  const float* yr = y + (size_t)row * 256;
  float v[8];
  float s = 0.f;
  #pragma unroll
  for (int i = 0; i < 8; ++i) { int c = lane + i * 32; v[i] = yr[c] + bias[c]; s += v[i]; }
  #pragma unroll
  for (int o = 16; o > 0; o >>= 1) s += __shfl_xor(s, o, 32);
  float mean = s * (1.f / 256.f);
  float q = 0.f;
  #pragma unroll
  for (int i = 0; i < 8; ++i) { float d = v[i] - mean; q += d * d; }
  #pragma unroll
  for (int o = 16; o > 0; o >>= 1) q += __shfl_xor(q, o, 32);
  float rstd = rsqrtf(q * (1.f / 256.f) + 1e-5f);
  #pragma unroll
  for (int i = 0; i < 8; ++i) {
    int c = lane + i * 32;
    float o = (v[i] - mean) * rstd * g[c] + b[c];
    if (do_relu) o = fmaxf(o, 0.f);
    if (res) o += res[(size_t)row * 256 + c];
    out[(size_t)row * 256 + c] = o;
  }
}

extern "C" void kernel_launch(void* const* d_in, const int* in_sizes, int n_in,
                              void* d_out, int out_size, void* d_ws, size_t ws_size,
                              hipStream_t stream) {
  (void)in_sizes; (void)n_in; (void)out_size; (void)ws_size;
  constexpr int NT = 80000, NP = 20000, N = 100000, E = 300000, H = 256, R = 5;
  constexpr int FT = 768, FP = 384;

  const float* thesis = (const float*)d_in[0];
  const float* prof   = (const float*)d_in[1];
  const int*   ei     = (const int*)d_in[2];
  const int*   et     = (const int*)d_in[3];
  const float* Wt  = (const float*)d_in[4];
  const float* bt  = (const float*)d_in[5];
  const float* ltg = (const float*)d_in[6];
  const float* ltb = (const float*)d_in[7];
  const float* Wp  = (const float*)d_in[8];
  const float* bp  = (const float*)d_in[9];
  const float* lpg = (const float*)d_in[10];
  const float* lpb = (const float*)d_in[11];
  const float* relw[3]  = {(const float*)d_in[12], (const float*)d_in[17], (const float*)d_in[22]};
  const float* rootw[3] = {(const float*)d_in[13], (const float*)d_in[18], (const float*)d_in[23]};
  const float* cb[3]    = {(const float*)d_in[14], (const float*)d_in[19], (const float*)d_in[24]};
  const float* lg[3]    = {(const float*)d_in[15], (const float*)d_in[20], (const float*)d_in[25]};
  const float* lb[3]    = {(const float*)d_in[16], (const float*)d_in[21], (const float*)d_in[26]};

  float* ws   = (float*)d_ws;
  float* sums = ws;                              // N*1280 (cnt must follow contiguously)
  float* cnt  = sums + (size_t)N * 1280;         // N*5
  float* ybuf = cnt  + (size_t)N * R;            // N*256
  float* xA   = ybuf + (size_t)N * H;            // N*256 (x1)
  float* xB   = xA   + (size_t)N * H;            // N*256 (x2)
  float* outAll = (float*)d_out;                 // all_features [N,256]
  float* outX3  = outAll + (size_t)N * H;        // x3 [N,256]
  const int* src = ei;
  const int* dst = ei + E;

  dim3 blk(256);

  // ---- input transforms: relu(LN(X @ W + b)) ----
  gemm_bf16_k<<<dim3(2, (NT + BM - 1) / BM), blk, 0, stream>>>(
      thesis, Wt, FT, nullptr, nullptr, 0, ybuf, NT);
  ln_k<<<dim3((NT + 7) / 8), blk, 0, stream>>>(ybuf, bt, ltg, ltb, nullptr, outAll, NT, 1);

  gemm_bf16_k<<<dim3(2, (NP + BM - 1) / BM), blk, 0, stream>>>(
      prof, Wp, FP, nullptr, nullptr, 0, ybuf, NP);
  ln_k<<<dim3((NP + 7) / 8), blk, 0, stream>>>(ybuf, bp, lpg, lpb, nullptr,
                                               outAll + (size_t)NT * H, NP, 1);

  // ---- 3 RGCN layers: mean-aggregate first (linearity), then one fused GEMM ----
  const float* xin = outAll;
  float* xouts[3] = {xA, xB, outX3};
  const int ztotal = N * 1280 + N * R;  // sums + cnt, contiguous, multiple of 4
  for (int L = 0; L < 3; ++L) {
    zero_k<<<dim3((ztotal / 4 + 255) / 256), blk, 0, stream>>>(sums, ztotal);
    scatter_k<<<dim3((E + 3) / 4), blk, 0, stream>>>(xin, src, dst, et, sums, cnt, E);
    meandiv_k<<<dim3((N * 1280 + 255) / 256), blk, 0, stream>>>(sums, cnt, N * 1280);
    // y = mean[N,1280] @ rel_w(flat [1280,256]) + x[N,256] @ root_w
    gemm_bf16_k<<<dim3(2, (N + BM - 1) / BM), blk, 0, stream>>>(
        sums, relw[L], 1280, xin, rootw[L], 256, ybuf, N);
    // x_next = [relu]( LN(y + cb) ) + x_prev   (no relu on layer 3)
    ln_k<<<dim3((N + 7) / 8), blk, 0, stream>>>(
        ybuf, cb[L], lg[L], lb[L], xin, xouts[L], N, (L < 2) ? 1 : 0);
    xin = xouts[L];
  }
}
